// Net_44805098832396
// MI455X (gfx1250) — compile-verified, matrix operand order checked by reading, not run on hardware
//
#include <hip/hip_runtime.h>

#define DI __device__ __forceinline__

typedef __attribute__((ext_vector_type(16))) __bf16        bf16x16;
typedef __attribute__((ext_vector_type(8)))  float         f32x8;
typedef __attribute__((ext_vector_type(4)))  float         f32x4;
typedef __attribute__((ext_vector_type(4)))  unsigned int  u32x4;

union FragB { bf16x16 v; u32x4 q[2]; };
union FragC { f32x8 v; f32x4 h[2]; float f[8]; };

constexpr int B_  = 256;
constexpr int T_  = 1024;
constexpr int INF = 9;
constexpr int CC  = 256;
constexpr int HID = 64;
constexpr int G3  = 192;      // 3*HID
constexpr int NBT = B_ / 16;  // 16 batch tiles

DI unsigned short f2bf(float f) {
  union { float f; unsigned u; } x; x.f = f;
  unsigned r = x.u + 0x7FFFu + ((x.u >> 16) & 1u);   // RNE
  return (unsigned short)(r >> 16);
}
DI float sigmoidf_(float x) { return 1.0f / (1.0f + __expf(-x)); }

DI f32x8 splat8(float v) {
  f32x8 r;
#pragma unroll
  for (int i = 0; i < 8; ++i) r[i] = v;
  return r;
}

// Load a 16-half bf16 WMMA fragment from two contiguous 16-byte runs.
DI bf16x16 load_frag(const unsigned short* p0, const unsigned short* p1) {
  FragB f;
  f.q[0] = *(const u32x4*)p0;
  f.q[1] = *(const u32x4*)p1;
  return f.v;
}

DI f32x8 wmma_bf16(bf16x16 a, bf16x16 b, f32x8 c) {
  return __builtin_amdgcn_wmma_f32_16x16x32_bf16(false, a, false, b, (short)0, c,
                                                 false, false);
}

// ---------------------------------------------------------------- K0: fp32->bf16
__global__ void k_cvt(const float* __restrict__ a, unsigned short* __restrict__ o, int n) {
  int i = blockIdx.x * blockDim.x + threadIdx.x;
  if (i < n) o[i] = f2bf(a[i]);
}

// -------------------------------------------- K1: conv + sigmoid + GRU0 input proj
// Grid: NBT*64 blocks (batch-tile x t-chunk of 16), 128 threads = 4 waves.
// xp0 stored in WMMA fragment order: (((bt*T + t)*12 + tile)*32 + lane)*8 + r
// Tiles 0..7 (r,z gates) get b_ih0 + b_hh0 folded in; tiles 8..11 (n gate) b_ih0 only.
__global__ __launch_bounds__(128) void k_conv_proj(
    const float* __restrict__ x, const float* __restrict__ conv_w,
    const float* __restrict__ conv_b, const float* __restrict__ b_ih0,
    const float* __restrict__ b_hh0, const unsigned short* __restrict__ wih0,
    float* __restrict__ xp0) {
  const int bt  = blockIdx.x & (NBT - 1);
  const int tc  = blockIdx.x >> 4;              // 64 chunks of 16 timesteps
  const int tid = threadIdx.x, lane = tid & 31, wv = tid >> 5;
  const int g = lane >> 4, nl = lane & 15;

  __shared__ float          s_cw[CC * INF];
  __shared__ float          s_cb[CC];
  __shared__ unsigned short s_hc[16 * CC];

  for (int i = tid; i < CC * INF; i += 128) s_cw[i] = conv_w[i];
  for (int i = tid; i < CC; i += 128) s_cb[i] = conv_b[i];

  // B fragments of w_ih0^T held in registers for the whole kernel (24 frags/wave).
  bf16x16 bfw[3][8];
  float   cinit[3];
  size_t  dstBase[3];
#pragma unroll
  for (int i = 0; i < 3; ++i) {
    const int tile = wv * 3 + i;
    const int n    = tile * 16 + nl;
    cinit[i]       = b_ih0[n] + (tile < 8 ? b_hh0[n] : 0.f);
    dstBase[i]     = (size_t)bt * T_ * 3072 + (size_t)tile * 256 + (size_t)lane * 8;
    const unsigned short* wr = wih0 + n * CC;
#pragma unroll
    for (int s = 0; s < 8; ++s)
      bfw[i][s] = load_frag(wr + 32 * s + 8 * g, wr + 32 * s + 16 + 8 * g);
  }
  __syncthreads();

  const int m = tid >> 3, c0 = (tid & 7) * 32;
  const float* xrow = x + ((size_t)(bt * 16 + m) * T_ + tc * 16) * INF;
  float xv[INF];
#pragma unroll
  for (int f = 0; f < INF; ++f) xv[f] = xrow[f];

  for (int tt = 0; tt < 16; ++tt) {
    const int t = tc * 16 + tt;
    // conv: 16 rows x 256 channels, K=9 on VALU
    for (int c = c0; c < c0 + 32; ++c) {
      float acc = s_cb[c];
#pragma unroll
      for (int f = 0; f < INF; ++f) acc += xv[f] * s_cw[c * INF + f];
      s_hc[m * CC + c] = f2bf(sigmoidf_(acc));
    }
    // prefetch next conv input row while WMMA phase runs
    float xvn[INF];
    if (tt + 1 < 16) {
#pragma unroll
      for (int f = 0; f < INF; ++f) xvn[f] = xrow[(tt + 1) * INF + f];
    }
    __syncthreads();

    // 16x192 = 12 tiles, K=256 = 8 k-steps; 24 WMMA per wave per t
    FragC acc[3];
#pragma unroll
    for (int i = 0; i < 3; ++i) acc[i].v = splat8(cinit[i]);
    const unsigned short* hr = s_hc + (lane & 15) * CC;
#pragma unroll
    for (int s = 0; s < 8; ++s) {
      bf16x16 a = load_frag(hr + 32 * s + 8 * g, hr + 32 * s + 16 + 8 * g);
#pragma unroll
      for (int i = 0; i < 3; ++i) acc[i].v = wmma_bf16(a, bfw[i][s], acc[i].v);
    }
#pragma unroll
    for (int i = 0; i < 3; ++i) {
      float* dst = xp0 + dstBase[i] + (size_t)t * 3072;
      *(f32x4*)dst       = acc[i].h[0];
      *(f32x4*)(dst + 4) = acc[i].h[1];
    }
    __syncthreads();
#pragma unroll
    for (int f = 0; f < INF; ++f) xv[f] = xvn[f];
  }
}

// ------------------------------------------------------------- K2: GRU layer 0
// Grid: NBT blocks (one per 16-row batch tile), 128 threads; all 1024 steps local.
// Software-pipelined: xp0 fragments / xn gathers for step t+1 are prefetched into a
// register double-buffer while step t computes (global latency off the critical path).
// y0 stored bf16 in A-fragment order: (((bt*T + t)*2 + kstep)*32 + lane)*16 + j
__global__ __launch_bounds__(128) void k_gru0(
    const float* __restrict__ b_hh0, const unsigned short* __restrict__ whh0,
    const float* __restrict__ xp0, unsigned short* __restrict__ y0) {
  const int bt  = blockIdx.x;
  const int tid = threadIdx.x, lane = tid & 31, wv = tid >> 5;
  const int g = lane >> 4, nl = lane & 15;

  __shared__ float          s_gh[16 * G3];
  __shared__ float          s_hf[16 * HID];
  __shared__ unsigned short s_hb[16 * HID];

  bf16x16 bh[3][2];
  float   bhh[3];
  size_t  xfBase[3];
#pragma unroll
  for (int i = 0; i < 3; ++i) {
    const int tile = wv * 3 + i;
    const int n    = tile * 16 + nl;
    bhh[i]         = b_hh0[n];
    xfBase[i]      = (size_t)bt * T_ * 3072 + (size_t)tile * 256 + (size_t)lane * 8;
    const unsigned short* wr = whh0 + n * HID;
#pragma unroll
    for (int s = 0; s < 2; ++s)
      bh[i][s] = load_frag(wr + 32 * s + 8 * g, wr + 32 * s + 16 + 8 * g);
  }
  for (int i = tid; i < 16 * HID; i += 128) { s_hf[i] = 0.f; s_hb[i] = 0; }

  // elementwise-phase constants: thread owns (row em, cols j0..j0+7)
  const int em = tid >> 3, j0 = (tid & 7) * 8;
  size_t xnBase[8];
#pragma unroll
  for (int jj = 0; jj < 8; ++jj) {
    const int j  = j0 + jj;
    const int tl = 8 + (j >> 4);
    const int ln = (j & 15) + ((em >> 3) << 4);
    xnBase[jj]   = (size_t)bt * T_ * 3072 + (size_t)tl * 256 + (size_t)ln * 8 + (em & 7);
  }
  const int ks = j0 >> 5, jk = j0 & 31, part = jk >> 4, gg = (jk & 15) >> 3;
  const size_t yBase =
      (size_t)bt * T_ * 1024 + ((size_t)(ks * 32 + gg * 16 + em)) * 16 + part * 8;

  // prologue prefetch (t = 0)
  FragC xfA[3], xfB[3];
  float xnA[8], xnB[8];
#pragma unroll
  for (int i = 0; i < 3; ++i)
    if (wv * 3 + i < 8) {
      const float* xf = xp0 + xfBase[i];
      xfA[i].h[0] = *(const f32x4*)xf;
      xfA[i].h[1] = *(const f32x4*)(xf + 4);
    }
#pragma unroll
  for (int jj = 0; jj < 8; ++jj) xnA[jj] = xp0[xnBase[jj]];
  __syncthreads();

  auto step = [&](int t, FragC (&xfc)[3], float (&xnc)[8], FragC (&xfn)[3],
                  float (&xnn)[8]) {
    // prefetch t+1 data first so it overlaps this step's compute
    if (t + 1 < T_) {
      const size_t toff = (size_t)(t + 1) * 3072;
#pragma unroll
      for (int i = 0; i < 3; ++i)
        if (wv * 3 + i < 8) {
          const float* xf = xp0 + xfBase[i] + toff;
          xfn[i].h[0] = *(const f32x4*)xf;
          xfn[i].h[1] = *(const f32x4*)(xf + 4);
        }
#pragma unroll
      for (int jj = 0; jj < 8; ++jj) xnn[jj] = xp0[xnBase[jj] + toff];
    }
    // A fragments of h (bf16) from LDS
    const unsigned short* hr = s_hb + (lane & 15) * HID;
    bf16x16 a0 = load_frag(hr + 8 * g, hr + 16 + 8 * g);
    bf16x16 a1 = load_frag(hr + 32 + 8 * g, hr + 48 + 8 * g);
#pragma unroll
    for (int i = 0; i < 3; ++i) {
      const int tile = wv * 3 + i;
      FragC c;
      if (tile < 8) c = xfc[i];         // xp already holds b_ih + b_hh
      else          c.v = splat8(bhh[i]);
      c.v = wmma_bf16(a0, bh[i][0], c.v);
      c.v = wmma_bf16(a1, bh[i][1], c.v);
#pragma unroll
      for (int r = 0; r < 8; ++r)
        s_gh[(r + 8 * g) * G3 + tile * 16 + nl] = c.f[r];
    }
    __syncthreads();

    unsigned short hb[8];
#pragma unroll
    for (int jj = 0; jj < 8; ++jj) {
      const int j = j0 + jj;
      float rv = sigmoidf_(s_gh[em * G3 + j]);
      float zv = sigmoidf_(s_gh[em * G3 + 64 + j]);
      float hn = s_gh[em * G3 + 128 + j];
      float nv = tanhf(xnc[jj] + rv * hn);
      float h  = (1.f - zv) * nv + zv * s_hf[em * HID + j];
      s_hf[em * HID + j] = h;
      unsigned short b   = f2bf(h);
      s_hb[em * HID + j] = b;
      hb[jj]             = b;
    }
    // y0 fragment-order store (one b128 per thread)
    unsigned short* yp = y0 + yBase + (size_t)t * 1024;
    u32x4 pk;
    pk[0] = (unsigned)hb[0] | ((unsigned)hb[1] << 16);
    pk[1] = (unsigned)hb[2] | ((unsigned)hb[3] << 16);
    pk[2] = (unsigned)hb[4] | ((unsigned)hb[5] << 16);
    pk[3] = (unsigned)hb[6] | ((unsigned)hb[7] << 16);
    *(u32x4*)yp = pk;
    __syncthreads();
  };

  for (int t = 0; t < T_; t += 2) {
    step(t, xfA, xnA, xfB, xnB);
    step(t + 1, xfB, xnB, xfA, xnA);
  }
}

// ------------------------------------------------------------- K3: GRU layer 1
// Fuses the input projection (y0 @ w_ih1^T) into the recurrence: 48 WMMA/step.
// y0 A-fragments for step t+1 prefetched into a register double-buffer.
__global__ __launch_bounds__(128) void k_gru1(
    const float* __restrict__ b_ih1, const float* __restrict__ b_hh1,
    const unsigned short* __restrict__ wih1, const unsigned short* __restrict__ whh1,
    const unsigned short* __restrict__ y0, float* __restrict__ h1f) {
  const int bt  = blockIdx.x;
  const int tid = threadIdx.x, lane = tid & 31, wv = tid >> 5;
  const int g = lane >> 4, nl = lane & 15;

  __shared__ float          s_gh[16 * G3];
  __shared__ float          s_xn[16 * HID];
  __shared__ float          s_hf[16 * HID];
  __shared__ unsigned short s_hb[16 * HID];

  bf16x16 bi[3][2], bh[3][2];
  float   bih[3], bhh[3];
#pragma unroll
  for (int i = 0; i < 3; ++i) {
    const int n = (wv * 3 + i) * 16 + nl;
    bih[i] = b_ih1[n];
    bhh[i] = b_hh1[n];
    const unsigned short* wi = wih1 + n * HID;
    const unsigned short* wh = whh1 + n * HID;
#pragma unroll
    for (int s = 0; s < 2; ++s) {
      bi[i][s] = load_frag(wi + 32 * s + 8 * g, wi + 32 * s + 16 + 8 * g);
      bh[i][s] = load_frag(wh + 32 * s + 8 * g, wh + 32 * s + 16 + 8 * g);
    }
  }
  for (int i = tid; i < 16 * HID; i += 128) { s_hf[i] = 0.f; s_hb[i] = 0; }

  const int em = tid >> 3, j0 = (tid & 7) * 8;
  const size_t yBase = (size_t)bt * T_ * 1024 + (size_t)lane * 16;

  // prologue prefetch (t = 0)
  bf16x16 yaA[2], yaB[2];
  {
    const unsigned short* yb = y0 + yBase;
    yaA[0] = load_frag(yb, yb + 8);
    yaA[1] = load_frag(yb + 512, yb + 512 + 8);
  }
  __syncthreads();

  auto step = [&](int t, bf16x16 (&yac)[2], bf16x16 (&yan)[2]) {
    if (t + 1 < T_) {
      const unsigned short* yb = y0 + yBase + (size_t)(t + 1) * 1024;
      yan[0] = load_frag(yb, yb + 8);
      yan[1] = load_frag(yb + 512, yb + 512 + 8);
    }
    const unsigned short* hr = s_hb + (lane & 15) * HID;
    bf16x16 a0 = load_frag(hr + 8 * g, hr + 16 + 8 * g);
    bf16x16 a1 = load_frag(hr + 32 + 8 * g, hr + 48 + 8 * g);
#pragma unroll
    for (int i = 0; i < 3; ++i) {
      const int tile = wv * 3 + i;
      if (tile < 8) {  // r,z: sum input + hidden pre-activations in one accumulator
        FragC c;
        c.v = splat8(bih[i] + bhh[i]);
        c.v = wmma_bf16(yac[0], bi[i][0], c.v);
        c.v = wmma_bf16(yac[1], bi[i][1], c.v);
        c.v = wmma_bf16(a0, bh[i][0], c.v);
        c.v = wmma_bf16(a1, bh[i][1], c.v);
#pragma unroll
        for (int r = 0; r < 8; ++r)
          s_gh[(r + 8 * g) * G3 + tile * 16 + nl] = c.f[r];
      } else {         // n: keep xn and hn separate (r scales only hn)
        FragC cx, ch;
        cx.v = splat8(bih[i]);
        cx.v = wmma_bf16(yac[0], bi[i][0], cx.v);
        cx.v = wmma_bf16(yac[1], bi[i][1], cx.v);
        ch.v = splat8(bhh[i]);
        ch.v = wmma_bf16(a0, bh[i][0], ch.v);
        ch.v = wmma_bf16(a1, bh[i][1], ch.v);
#pragma unroll
        for (int r = 0; r < 8; ++r) {
          s_gh[(r + 8 * g) * G3 + tile * 16 + nl]        = ch.f[r];
          s_xn[(r + 8 * g) * HID + (tile - 8) * 16 + nl] = cx.f[r];
        }
      }
    }
    __syncthreads();
#pragma unroll
    for (int jj = 0; jj < 8; ++jj) {
      const int j = j0 + jj;
      float rv = sigmoidf_(s_gh[em * G3 + j]);
      float zv = sigmoidf_(s_gh[em * G3 + 64 + j]);
      float hn = s_gh[em * G3 + 128 + j];
      float xn = s_xn[em * HID + j];
      float nv = tanhf(xn + rv * hn);
      float h  = (1.f - zv) * nv + zv * s_hf[em * HID + j];
      s_hf[em * HID + j] = h;
      s_hb[em * HID + j] = f2bf(h);
      if (t == T_ - 1) h1f[(size_t)(bt * 16 + em) * HID + j] = h;
    }
    __syncthreads();
  };

  for (int t = 0; t < T_; t += 2) {
    step(t, yaA, yaB);
    step(t + 1, yaB, yaA);
  }
}

// ---------------------------------------------------------------- K4: final FC
__global__ void k_fc(const float* __restrict__ h1f, const float* __restrict__ fc_w,
                     const float* __restrict__ fc_b, float* __restrict__ out) {
  int b = blockIdx.x * blockDim.x + threadIdx.x;
  if (b < B_) {
    float acc = fc_b[0];
#pragma unroll
    for (int j = 0; j < HID; ++j) acc += h1f[(size_t)b * HID + j] * fc_w[j];
    out[b] = acc;
  }
}

extern "C" void kernel_launch(void* const* d_in, const int* in_sizes, int n_in,
                              void* d_out, int out_size, void* d_ws, size_t ws_size,
                              hipStream_t stream) {
  (void)in_sizes; (void)n_in; (void)out_size; (void)ws_size;
  const float* x      = (const float*)d_in[0];
  const float* conv_w = (const float*)d_in[1];
  const float* conv_b = (const float*)d_in[2];
  const float* w_ih0  = (const float*)d_in[3];
  const float* w_hh0  = (const float*)d_in[4];
  const float* b_ih0  = (const float*)d_in[5];
  const float* b_hh0  = (const float*)d_in[6];
  const float* w_ih1  = (const float*)d_in[7];
  const float* w_hh1  = (const float*)d_in[8];
  const float* b_ih1  = (const float*)d_in[9];
  const float* b_hh1  = (const float*)d_in[10];
  const float* fc_w   = (const float*)d_in[11];
  const float* fc_b   = (const float*)d_in[12];
  float* out = (float*)d_out;

  char* ws = (char*)d_ws;
  const size_t XP0B = (size_t)B_ * T_ * G3 * 4;   // 201 MB, gate pre-activations
  const size_t Y0B  = (size_t)B_ * T_ * HID * 2;  // 33.5 MB, GRU0 outputs (bf16)
  const size_t H1B  = (size_t)B_ * HID * 4;
  float*          xp0   = (float*)ws;
  unsigned short* y0    = (unsigned short*)(ws + XP0B);
  float*          h1f   = (float*)(ws + XP0B + Y0B);
  unsigned short* wih0b = (unsigned short*)(ws + XP0B + Y0B + H1B);
  unsigned short* whh0b = wih0b + G3 * CC;
  unsigned short* wih1b = whh0b + G3 * HID;
  unsigned short* whh1b = wih1b + G3 * HID;

  k_cvt<<<(G3 * CC + 255) / 256, 256, 0, stream>>>(w_ih0, wih0b, G3 * CC);
  k_cvt<<<(G3 * HID + 255) / 256, 256, 0, stream>>>(w_hh0, whh0b, G3 * HID);
  k_cvt<<<(G3 * HID + 255) / 256, 256, 0, stream>>>(w_ih1, wih1b, G3 * HID);
  k_cvt<<<(G3 * HID + 255) / 256, 256, 0, stream>>>(w_hh1, whh1b, G3 * HID);

  k_conv_proj<<<NBT * 64, 128, 0, stream>>>(x, conv_w, conv_b, b_ih0, b_hh0, wih0b, xp0);
  k_gru0<<<NBT, 128, 0, stream>>>(b_hh0, whh0b, xp0, y0);
  k_gru1<<<NBT, 128, 0, stream>>>(b_ih1, b_hh1, wih1b, whh1b, y0, h1f);
  k_fc<<<1, 256, 0, stream>>>(h1f, fc_w, fc_b, out);
}
